// GCN_decay_rep1_35871566856583
// MI455X (gfx1250) — matrix operand order repaired and checked by reading, not compiled, not run on hardware
//
#include <hip/hip_runtime.h>
#include <hip/hip_bf16.h>
#include <stdint.h>

typedef __attribute__((ext_vector_type(16))) _Float16 v16h;
typedef __attribute__((ext_vector_type(8)))  float    v8f;
typedef __attribute__((ext_vector_type(4)))  int      v4i;

#define N_NODES 50000
#define M_PAD   50048   // 391 * 128 : every A-side activation buffer padded to this
#define N_EDGES 400000
#define NFEAT   500
#define KPAD_X  512
#define NHID4   64
#define NCLASS  40

#ifndef __has_builtin
#define __has_builtin(x) 0
#endif

#if __has_builtin(__builtin_amdgcn_global_load_async_to_lds_b128)
#define HAVE_ASYNC_LDS 1
#else
#define HAVE_ASYNC_LDS 0
#endif

#if __has_builtin(__builtin_amdgcn_s_wait_asynccnt)
#define WAIT_ASYNC(n) __builtin_amdgcn_s_wait_asynccnt(n)
#else
#define WAIT_ASYNC(n) asm volatile("s_wait_asynccnt %0" ::"n"(n))
#endif

typedef __attribute__((address_space(1))) v4i* v4i_as1;
typedef __attribute__((address_space(3))) v4i* v4i_as3;

// ---------------------------------------------------------------------------
// WMMA GEMM: C[M,N] = A[M,K](f16,row-major,lda) * B, B pre-transposed as
// Bt[Npad][Kpad] (f16, ldb=Kpad, zero-padded). Block = 128 thr (4 waves),
// 128x64 C tile per block; each wave owns 32 rows x 64 cols = 8 accumulators
// (two A fragments share each B fragment -> 8 WMMA : 12 ds_load_b128).
// Double-buffered LDS tiles staged with GLOBAL_LOAD_ASYNC_TO_LDS_B128
// (ASYNCcnt) when available. A must be readable for gridDim.y*128 rows
// (buffers padded to M_PAD); garbage pad rows only feed unstored C rows.
// ---------------------------------------------------------------------------
__global__ __launch_bounds__(128)
void gemm_f16_wmma(const _Float16* __restrict__ A, int lda,
                   const _Float16* __restrict__ Bt, int ldb,
                   float* __restrict__ C, int ldc,
                   _Float16* __restrict__ C16, int ldc16,
                   const float* __restrict__ bias,
                   int M, int N, int K, int do_relu)
{
  __shared__ _Float16 As[2][128 * 32];   // 8 KB per buffer
  __shared__ _Float16 Bs[2][64 * 32];    // 4 KB per buffer

  const int tid  = threadIdx.x;
  const int wave = tid >> 5;
  const int lane = tid & 31;
  const int grp  = lane >> 4;            // half-wave group (ISA fragment layout)
  const int l16  = lane & 15;
  const long m0  = (long)blockIdx.y * 128;
  const long n0  = (long)blockIdx.x * 64;

  v8f acc[2][4] = {};

  auto issue_tile = [&](int buf, int kk) {
#if HAVE_ASYNC_LDS
#pragma unroll
    for (int i = 0; i < 4; ++i) {               // A: 512 x 16B chunks
      int c = i * 128 + tid;
      int r = c >> 2, col = (c & 3) * 8;
      const v4i* gp = (const v4i*)(A + (size_t)(m0 + r) * lda + kk + col);
      v4i*       lp = (v4i*)(&As[buf][r * 32 + col]);
      __builtin_amdgcn_global_load_async_to_lds_b128((v4i_as1)gp, (v4i_as3)lp, 0, 0);
    }
#pragma unroll
    for (int i = 0; i < 2; ++i) {               // B: 256 x 16B chunks
      int c = i * 128 + tid;
      int r = c >> 2, col = (c & 3) * 8;
      const v4i* gp = (const v4i*)(Bt + (size_t)(n0 + r) * ldb + kk + col);
      v4i*       lp = (v4i*)(&Bs[buf][r * 32 + col]);
      __builtin_amdgcn_global_load_async_to_lds_b128((v4i_as1)gp, (v4i_as3)lp, 0, 0);
    }
#else
#pragma unroll
    for (int i = 0; i < 4; ++i) {
      int c = i * 128 + tid;
      int r = c >> 2, col = (c & 3) * 8;
      *(v4i*)(&As[buf][r * 32 + col]) =
          *(const v4i*)(A + (size_t)(m0 + r) * lda + kk + col);
    }
#pragma unroll
    for (int i = 0; i < 2; ++i) {
      int c = i * 128 + tid;
      int r = c >> 2, col = (c & 3) * 8;
      *(v4i*)(&Bs[buf][r * 32 + col]) =
          *(const v4i*)(Bt + (size_t)(n0 + r) * ldb + kk + col);
    }
#endif
  };

  const int nk = K >> 5;
  issue_tile(0, 0);

  for (int ki = 0; ki < nk; ++ki) {
    const int buf = ki & 1;
    if (ki + 1 < nk) {                    // prefetch next K-slab into other buf
      issue_tile(buf ^ 1, (ki + 1) << 5);
#if HAVE_ASYNC_LDS
      WAIT_ASYNC(6);                      // 6 in flight for next tile; this tile done
#endif
    } else {
#if HAVE_ASYNC_LDS
      WAIT_ASYNC(0);
#endif
    }
    __syncthreads();                      // all waves' tile data visible

    // A fragments (16x32 f16 each): lanes 0-15 = M rows; VGPR v holds K pairs
    // v<4 -> K=2v(+grp*8), v>=4 -> K=16+2(v-4)(+grp*8).
    union { v16h h; unsigned u[8]; } a0, a1;
    const unsigned* ar0 = (const unsigned*)(&As[buf][(wave * 32 + l16) * 32]);
    const unsigned* ar1 = (const unsigned*)(&As[buf][(wave * 32 + 16 + l16) * 32]);
#pragma unroll
    for (int v = 0; v < 8; ++v) {
      int kb = (v < 4 ? v * 2 : 16 + (v - 4) * 2) + grp * 8;
      a0.u[v] = ar0[kb >> 1];
      a1.u[v] = ar1[kb >> 1];
    }
#pragma unroll
    for (int t = 0; t < 4; ++t) {
      // B fragment (32x16): lane%16 = N col; same K pairing as A fragments
      union { v16h h; unsigned u[8]; } b;
      const unsigned* br = (const unsigned*)(&Bs[buf][(t * 16 + l16) * 32]);
#pragma unroll
      for (int v = 0; v < 8; ++v) b.u[v] = br[((v < 4 ? v * 2 : 16 + (v - 4) * 2) + grp * 8) >> 1];
      acc[0][t] = __builtin_amdgcn_wmma_f32_16x16x32_f16(
          false, a0.h, false, b.h, (short)0, acc[0][t], false, false);
      acc[1][t] = __builtin_amdgcn_wmma_f32_16x16x32_f16(
          false, a1.h, false, b.h, (short)0, acc[1][t], false, false);
    }
    __syncthreads();                      // buf free for the next async write
  }

  // C/D layout: VGPR v, lane -> row = 8*grp + v, col = lane%16 per 16x16 tile
#pragma unroll
  for (int f = 0; f < 2; ++f) {
#pragma unroll
    for (int t = 0; t < 4; ++t) {
      long col = n0 + t * 16 + l16;
      float bv = 0.0f;
      if (bias != nullptr && col < N) bv = bias[col];
#pragma unroll
      for (int v = 0; v < 8; ++v) {
        long row = m0 + wave * 32 + f * 16 + grp * 8 + v;
        if (row < M && col < N) {
          float xv = acc[f][t][v] + bv;
          if (do_relu) xv = fmaxf(xv, 0.0f);
          if (C)   C[row * ldc + col] = xv;
          if (C16) C16[row * ldc16 + col] = (_Float16)xv;
        }
      }
    }
  }
}

// ---------------------------------------------------------------------------
// Data-prep / SpMM kernels
// ---------------------------------------------------------------------------
__global__ void k_cvt_x(const float* __restrict__ x, _Float16* __restrict__ xh) {
  int idx = blockIdx.x * blockDim.x + threadIdx.x;
  if (idx >= N_NODES * KPAD_X) return;
  int n = idx >> 9, c = idx & (KPAD_X - 1);
  float v = (c < NFEAT) ? x[(size_t)n * NFEAT + c] : 0.0f;
  xh[idx] = (_Float16)v;
}

// W [K][N] row-major (f32) -> Wt [Npad][Kpad] (f16), zero padded.
__global__ void k_cvt_wt(const float* __restrict__ W, _Float16* __restrict__ Wt,
                         int K, int N, int Kpad, int Npad) {
  int idx = blockIdx.x * blockDim.x + threadIdx.x;
  if (idx >= Npad * Kpad) return;
  int n = idx / Kpad, k = idx - n * Kpad;
  float v = (n < N && k < K) ? W[(size_t)k * N + n] : 0.0f;
  Wt[idx] = (_Float16)v;
}

__global__ void k_init_bias(float* __restrict__ out, const float* __restrict__ b,
                            int F, int total) {
  int idx = blockIdx.x * blockDim.x + threadIdx.x;
  if (idx >= total) return;
  out[idx] = b[idx % F];
}

// msg = support[src] * val scattered into out[dst] (+bias already in out).
// Consecutive lanes share an edge, stride features -> coalesced gather;
// atomics land in an L2-resident (25.6 MB << 192 MB) accumulator.
__global__ void k_spmm(const float* __restrict__ sup, const int* __restrict__ src,
                       const int* __restrict__ dst, const float* __restrict__ val,
                       float* __restrict__ out, int F, int total) {
  int idx = blockIdx.x * blockDim.x + threadIdx.x;
  if (idx >= total) return;
  int e = idx / F, f = idx - e * F;
  float m = sup[(size_t)src[e] * F + f] * val[e];
  atomicAdd(out + (size_t)dst[e] * F + f, m);
}

__global__ void k_relu_f16(const float* __restrict__ in, _Float16* __restrict__ oh,
                           int total) {
  int idx = blockIdx.x * blockDim.x + threadIdx.x;
  if (idx >= total) return;
  oh[idx] = (_Float16)fmaxf(in[idx], 0.0f);
}

// Branch-k output: pre-ReLU f16 concat (ensemble input) + ReLU'd f16 (heads),
// both [M_PAD][256] column blocks.
__global__ void k_xb_cvt(const float* __restrict__ acc64, _Float16* __restrict__ xb,
                         _Float16* __restrict__ xbr, int kbr) {
  int idx = blockIdx.x * blockDim.x + threadIdx.x;
  if (idx >= N_NODES * NHID4) return;
  int n = idx >> 6, j = idx & 63;
  float v = acc64[idx];
  size_t o = (size_t)n * 256 + kbr * 64 + j;
  xb[o]  = (_Float16)v;
  xbr[o] = (_Float16)fmaxf(v, 0.0f);
}

// ---------------------------------------------------------------------------
// Host orchestration
// ---------------------------------------------------------------------------
static inline size_t alignup256(size_t x) { return (x + 255) & ~(size_t)255; }

extern "C" void kernel_launch(void* const* d_in, const int* in_sizes, int n_in,
                              void* d_out, int out_size, void* d_ws, size_t ws_size,
                              hipStream_t stream) {
  (void)in_sizes; (void)n_in; (void)out_size; (void)ws_size;

  const float* x   = (const float*)d_in[0];
  const int*   src = (const int*)d_in[1];
  const int*   dst = (const int*)d_in[2];
  const float* val = (const float*)d_in[3];
  const float* Wa  = (const float*)d_in[4];
  const float* ba  = (const float*)d_in[5];
  const float* Wb  = (const float*)d_in[6];
  const float* bb  = (const float*)d_in[7];
  const float* Wc  = (const float*)d_in[8];
  const float* bc  = (const float*)d_in[9];
  const float* dW  = (const float*)d_in[10];
  const float* db  = (const float*)d_in[11];
  const float* e1W = (const float*)d_in[12];
  const float* e1b = (const float*)d_in[13];
  const float* e2W = (const float*)d_in[14];
  const float* e2b = (const float*)d_in[15];
  const float* e3W = (const float*)d_in[16];
  const float* e3b = (const float*)d_in[17];
  float* out = (float*)d_out;

  char* p = (char*)d_ws;
  auto carve = [&](size_t bytes) { char* r = p; p += alignup256(bytes); return r; };
  // A-side activation buffers padded to M_PAD rows (branch-free tile loads).
  _Float16* xh   = (_Float16*)carve((size_t)M_PAD * KPAD_X * 2);
  _Float16* WaT  = (_Float16*)carve((size_t)4 * 128 * 512 * 2);
  _Float16* WbT  = (_Float16*)carve((size_t)4 * 128 * 128 * 2);
  _Float16* WcT  = (_Float16*)carve((size_t)4 * 64 * 128 * 2);
  _Float16* dWT  = (_Float16*)carve((size_t)5 * 64 * 64 * 2);
  _Float16* e1WT = (_Float16*)carve((size_t)128 * 256 * 2);
  _Float16* e2WT = (_Float16*)carve((size_t)128 * 128 * 2);
  _Float16* e3WT = (_Float16*)carve((size_t)64 * 128 * 2);
  float*    sup  = (float*)carve((size_t)M_PAD * 128 * 4);
  float*    acc  = (float*)carve((size_t)M_PAD * 128 * 4);
  _Float16* hh   = (_Float16*)carve((size_t)M_PAD * 128 * 2);
  _Float16* xbh  = (_Float16*)carve((size_t)M_PAD * 256 * 2);
  _Float16* xbrh = (_Float16*)carve((size_t)M_PAD * 256 * 2);

  auto g1 = [](long long n) { return dim3((unsigned)((n + 255) / 256)); };
  auto gemm = [&](const _Float16* A, int lda, const _Float16* Bt, int ldb,
                  float* C, int ldc, _Float16* C16, int ldc16,
                  const float* bias, int M, int N, int Npad, int K, int relu) {
    dim3 grid(Npad / 64, (M + 127) / 128);
    gemm_f16_wmma<<<grid, 128, 0, stream>>>(A, lda, Bt, ldb, C, ldc, C16, ldc16,
                                            bias, M, N, K, relu);
  };

  // ---- precision staging: x and all weights -> f16 (weights transposed) ----
  k_cvt_x<<<g1((long long)N_NODES * KPAD_X), 256, 0, stream>>>(x, xh);
  for (int k = 0; k < 4; ++k) {
    k_cvt_wt<<<g1(128 * 512), 256, 0, stream>>>(Wa + (size_t)k * NFEAT * 128,
                                                WaT + (size_t)k * 128 * 512,
                                                NFEAT, 128, 512, 128);
    k_cvt_wt<<<g1(128 * 128), 256, 0, stream>>>(Wb + (size_t)k * 128 * 128,
                                                WbT + (size_t)k * 128 * 128,
                                                128, 128, 128, 128);
    k_cvt_wt<<<g1(64 * 128), 256, 0, stream>>>(Wc + (size_t)k * 128 * 64,
                                               WcT + (size_t)k * 64 * 128,
                                               128, 64, 128, 64);
  }
  for (int k = 0; k < 5; ++k)
    k_cvt_wt<<<g1(64 * 64), 256, 0, stream>>>(dW + (size_t)k * 64 * NCLASS,
                                              dWT + (size_t)k * 64 * 64,
                                              64, NCLASS, 64, 64);
  k_cvt_wt<<<g1(128 * 256), 256, 0, stream>>>(e1W, e1WT, 256, 128, 256, 128);
  k_cvt_wt<<<g1(128 * 128), 256, 0, stream>>>(e2W, e2WT, 128, 128, 128, 128);
  k_cvt_wt<<<g1(64 * 128), 256, 0, stream>>>(e3W, e3WT, 128, 64, 128, 64);

  // ---- 4 GCN branches ----
  for (int k = 0; k < 4; ++k) {
    const int*   sk = src + (size_t)k * N_EDGES;
    const int*   dk = dst + (size_t)k * N_EDGES;
    const float* vk = val + (size_t)k * N_EDGES;

    // layer A: support = x @ Wa_k ; acc = bias ; scatter ; relu->f16
    gemm(xh, KPAD_X, WaT + (size_t)k * 128 * 512, 512,
         sup, 128, nullptr, 0, nullptr, N_NODES, 128, 128, 512, 0);
    k_init_bias<<<g1((long long)N_NODES * 128), 256, 0, stream>>>(acc, ba + k * 128, 128, N_NODES * 128);
    k_spmm<<<g1((long long)N_EDGES * 128), 256, 0, stream>>>(sup, sk, dk, vk, acc, 128, N_EDGES * 128);
    k_relu_f16<<<g1((long long)N_NODES * 128), 256, 0, stream>>>(acc, hh, N_NODES * 128);

    // layer B
    gemm(hh, 128, WbT + (size_t)k * 128 * 128, 128,
         sup, 128, nullptr, 0, nullptr, N_NODES, 128, 128, 128, 0);
    k_init_bias<<<g1((long long)N_NODES * 128), 256, 0, stream>>>(acc, bb + k * 128, 128, N_NODES * 128);
    k_spmm<<<g1((long long)N_EDGES * 128), 256, 0, stream>>>(sup, sk, dk, vk, acc, 128, N_EDGES * 128);
    k_relu_f16<<<g1((long long)N_NODES * 128), 256, 0, stream>>>(acc, hh, N_NODES * 128);

    // layer C (pre-ReLU branch output)
    gemm(hh, 128, WcT + (size_t)k * 64 * 128, 128,
         sup, 64, nullptr, 0, nullptr, N_NODES, 64, 64, 128, 0);
    k_init_bias<<<g1((long long)N_NODES * 64), 256, 0, stream>>>(acc, bc + k * 64, 64, N_NODES * 64);
    k_spmm<<<g1((long long)N_EDGES * 64), 256, 0, stream>>>(sup, sk, dk, vk, acc, 64, N_EDGES * 64);
    k_xb_cvt<<<g1((long long)N_NODES * 64), 256, 0, stream>>>(acc, xbh, xbrh, k);

    // per-branch dense head: relu(xb_k) @ dW_k + db_k -> out[k]
    gemm(xbrh + k * 64, 256, dWT + (size_t)k * 64 * 64, 64,
         out + (size_t)k * N_NODES * NCLASS, NCLASS, nullptr, 0,
         db + k * NCLASS, N_NODES, NCLASS, 64, 64, 0);
  }

  // ---- ensemble MLP on pre-ReLU concat (reuse freed branch buffers) ----
  _Float16* e1o = hh;              // [M_PAD,128] f16
  _Float16* e2o = (_Float16*)sup;  // [M_PAD,128] f16 (fits in 25.6 MB f32 buf)
  _Float16* e3o = (_Float16*)acc;  // [M_PAD,64]  f16
  gemm(xbh, 256, e1WT, 256, nullptr, 0, e1o, 128, e1b, N_NODES, 128, 128, 256, 1);
  gemm(e1o, 128, e2WT, 128, nullptr, 0, e2o, 128, e2b, N_NODES, 128, 128, 128, 1);
  gemm(e2o, 128, e3WT, 128, nullptr, 0, e3o, 64, e3b, N_NODES, 64, 64, 128, 1);
  gemm(e3o, 64, dWT + (size_t)4 * 64 * 64, 64,
       out + (size_t)4 * N_NODES * NCLASS, NCLASS, nullptr, 0,
       db + 4 * NCLASS, N_NODES, NCLASS, 64, 64, 0);
}
